// DualGCNModel_23845658427619
// MI455X (gfx1250) — matrix operand order repaired and checked by reading, not compiled, not run on hardware
//
#include <hip/hip_runtime.h>
#include <hip/hip_bf16.h>

#define N_NODES 100000

typedef __attribute__((ext_vector_type(16))) _Float16 v16h;
typedef __attribute__((ext_vector_type(8)))  _Float16 v8h;
typedef __attribute__((ext_vector_type(8)))  float    v8f;
typedef __attribute__((ext_vector_type(4)))  float    v4f;

// ---------------------------------------------------------------- elementwise
__global__ void k_fill(float* __restrict__ p, float v, long long n) {
  long long i = (long long)blockIdx.x * blockDim.x + threadIdx.x;
  long long stride = (long long)gridDim.x * blockDim.x;
  for (; i < n; i += stride) p[i] = v;
}

__global__ void k_degrees(const int* __restrict__ src, const int* __restrict__ dst,
                          float* __restrict__ outdeg, float* __restrict__ indeg, int e) {
  int i = blockIdx.x * blockDim.x + threadIdx.x;
  int stride = gridDim.x * blockDim.x;
  for (; i < e; i += stride) {
    atomicAdd(&outdeg[src[i]], 1.0f);
    atomicAdd(&indeg[dst[i]], 1.0f);
  }
}

__global__ void k_rsqrt_clamp(float* __restrict__ p, int n) {
  int i = blockIdx.x * blockDim.x + threadIdx.x;
  if (i < n) p[i] = rsqrtf(fmaxf(p[i], 1.0f));
}

__global__ void k_rowscale(const float* __restrict__ x, const float* __restrict__ s,
                           float* __restrict__ y, int n, int w) {
  long long total = (long long)n * w;
  long long i = (long long)blockIdx.x * blockDim.x + threadIdx.x;
  long long stride = (long long)gridDim.x * blockDim.x;
  for (; i < total; i += stride) y[i] = x[i] * s[i / w];
}

// epilogue after aggregation: out[row, coff+col] = in[row,col]*rs[row] + bias[col]
__global__ void k_scale_bias(const float* __restrict__ in, const float* __restrict__ rs,
                             const float* __restrict__ bias, float* __restrict__ out,
                             int n, int w, int ld, int coff) {
  long long total = (long long)n * w;
  long long i = (long long)blockIdx.x * blockDim.x + threadIdx.x;
  long long stride = (long long)gridDim.x * blockDim.x;
  for (; i < total; i += stride) {
    int row = (int)(i / w);
    int col = (int)(i % w);
    out[(long long)row * ld + coff + col] = in[i] * rs[row] + bias[col];
  }
}

// W[K,N] f32 (row-major) -> Wt[N,K] f16 (each output row = one W column, contiguous in K)
__global__ void k_w_to_f16t(const float* __restrict__ W, _Float16* __restrict__ Wt,
                            int K, int N) {
  int total = K * N;
  int i = blockIdx.x * blockDim.x + threadIdx.x;
  int stride = gridDim.x * blockDim.x;
  for (; i < total; i += stride) {
    int k = i / N, nn = i % N;
    Wt[nn * K + k] = (_Float16)W[i];
  }
}

// ---------------------------------------------------------------- aggregation
// out[dst[e], :] += x[src[e], :]  (edge-parallel, float4 gather + f32 atomics;
// node matrices fit in the 192MB L2, so the random traffic stays on-chip)
__global__ void k_aggregate(const int* __restrict__ src, const int* __restrict__ dst,
                            const float* __restrict__ x, float* __restrict__ out,
                            int e, int w) {
  int chunks = w >> 2;                      // float4 chunks per row
  long long total = (long long)e * chunks;
  long long i = (long long)blockIdx.x * blockDim.x + threadIdx.x;
  long long stride = (long long)gridDim.x * blockDim.x;
  for (; i < total; i += stride) {
    long long edge = i / chunks;
    int c = (int)(i % chunks) << 2;
    int s = src[edge];
    int d = dst[edge];
    const v4f v = *(const v4f*)(x + (long long)s * w + c);
    float* o = out + (long long)d * w + c;
    atomicAdd(o + 0, v[0]);
    atomicAdd(o + 1, v[1]);
    atomicAdd(o + 2, v[2]);
    atomicAdd(o + 3, v[3]);
  }
}

// ---------------------------------------------------------------- WMMA GEMM
// C[M,N] = op(A[M,K]) @ W[K,N]; one wave32 per 16x64 tile (4 accumulators,
// A fragment reused across 4 WMMAs per k-step). Weights are pre-transposed
// f16 (Wt[N][K]); all base pointers hoisted so each fragment load is a
// b128 off a stable base register with a constant immediate offset.
// Fragment layouts per CDNA5 ISA 7.12.2 (wave32):
//   A 16x32 f16 : lane (g=lane/16, m=lane%16), elem i -> K = (i/8)*16 + g*8 + (i%8)
//                 => two contiguous runs of 8 floats: [k0+g*8, +8) and [k0+16+g*8, +8)
//   B 32x16 f16 : lane (g, n=lane%16),         elem i -> K = (i/8)*16 + g*8 + (i%8)
//                 => two contiguous runs of 8 halves at Wt[n][k0+g*8] and Wt[n][k0+16+g*8]
//   C 16x16 f32 : vgpr j, lane (g, n)          -> M = j + 8*g
template <int K, bool PRE, bool POST, bool BIAS, bool RELU>
__global__ void k_gemm_wmma(const float* __restrict__ A, const _Float16* __restrict__ Wt,
                            const float* __restrict__ bias,
                            const float* __restrict__ pre_scale,
                            const float* __restrict__ post_scale,
                            float* __restrict__ C,
                            int M, int N) {
  constexpr int TN = 4;                     // 4 x 16 = 64 output columns per wave
  int wave = blockIdx.x * (blockDim.x >> 5) + (threadIdx.x >> 5);
  int lane = threadIdx.x & 31;
  int nblk = N >> 6;                        // 64-column blocks
  int total = (M >> 4) * nblk;
  if (wave >= total) return;
  int tm  = wave / nblk;
  int tnb = wave % nblk;
  int g = lane >> 4;
  int r = lane & 15;
  int m   = (tm << 4) + r;                  // A row handled by this lane
  int nn0 = (tnb << 6) + r;                 // first B/C column handled by this lane

  float s = PRE ? pre_scale[m] : 1.0f;

  // Loop-invariant base pointers (k0 becomes a compile-time immediate offset).
  const v4f* ap0 = (const v4f*)(A + (long long)m * K + (g << 3));        // run 1
  const v4f* ap1 = (const v4f*)(A + (long long)m * K + 16 + (g << 3));   // run 2
  const v8h* wp[TN];
#pragma unroll
  for (int t = 0; t < TN; ++t)
    wp[t] = (const v8h*)(Wt + (long long)(nn0 + 16 * t) * K + (g << 3));

  v8f acc[TN] = {};
#pragma unroll
  for (int k0 = 0; k0 < K; k0 += 32) {
    // ---- A fragment: 4 x b128 loads of f32 (immediate offsets), scale+convert
    v4f a0 = ap0[k0 >> 2], a1 = ap0[(k0 >> 2) + 1];
    v4f a2 = ap1[k0 >> 2], a3 = ap1[(k0 >> 2) + 1];
    v16h a;
#pragma unroll
    for (int i = 0; i < 4; ++i) {
      a[i]      = (_Float16)(a0[i] * s);
      a[i + 4]  = (_Float16)(a1[i] * s);
      a[i + 8]  = (_Float16)(a2[i] * s);
      a[i + 12] = (_Float16)(a3[i] * s);
    }
#pragma unroll
    for (int t = 0; t < TN; ++t) {
      // ---- B fragment: 2 x b128 loads of contiguous f16 (immediate offsets)
      v8h blo = wp[t][k0 >> 3];
      v8h bhi = wp[t][(k0 >> 3) + 2];
      v16h b;
#pragma unroll
      for (int i = 0; i < 8; ++i) { b[i] = blo[i]; b[i + 8] = bhi[i]; }
      acc[t] = __builtin_amdgcn_wmma_f32_16x16x32_f16(
          false, a, false, b, (short)0, acc[t], false, false);
    }
  }

  // branch-free epilogue; post_scale loads hoisted (shared across the 4 subtiles)
  float ps[8];
#pragma unroll
  for (int j = 0; j < 8; ++j)
    ps[j] = POST ? post_scale[(tm << 4) + (g << 3) + j] : 1.0f;

#pragma unroll
  for (int t = 0; t < TN; ++t) {
    int nn = nn0 + 16 * t;
    float bv = BIAS ? bias[nn] : 0.0f;
#pragma unroll
    for (int j = 0; j < 8; ++j) {
      int mm = (tm << 4) + (g << 3) + j;
      float v = acc[t][j];
      if (POST) v *= ps[j];
      if (BIAS) v += bv;
      if (RELU) v = fmaxf(v, 0.0f);
      C[(long long)mm * N + nn] = v;
    }
  }
}

// ---------------------------------------------------------------- launcher
extern "C" void kernel_launch(void* const* d_in, const int* in_sizes, int n_in,
                              void* d_out, int out_size, void* d_ws, size_t ws_size,
                              hipStream_t stream) {
  const int n = N_NODES;
  const int e = in_sizes[0];

  const int*   src  = (const int*)d_in[0];
  const int*   dst  = (const int*)d_in[1];
  const float* ori  = (const float*)d_in[2];
  const float* strc = (const float*)d_in[3];
  const float* W1o = (const float*)d_in[4];  const float* b1o = (const float*)d_in[5];
  const float* W2o = (const float*)d_in[6];  const float* b2o = (const float*)d_in[7];
  const float* W1s = (const float*)d_in[8];  const float* b1s = (const float*)d_in[9];
  const float* W2s = (const float*)d_in[10]; const float* b2s = (const float*)d_in[11];
  const float* Wm1 = (const float*)d_in[12]; const float* bm1 = (const float*)d_in[13];
  const float* Wm2 = (const float*)d_in[14]; const float* bm2 = (const float*)d_in[15];
  float* out = (float*)d_out;

  // workspace layout (floats): 2n + 128n + 128n + 128n + 64n + 128n = 578n (~231 MB)
  // then ~90K f16 for the transposed weights
  float* ws   = (float*)d_ws;
  float* dois = ws;                       // deg_out^-1/2  [n]
  float* diis = ws + n;                   // deg_in^-1/2   [n]
  float* bufA = ws + 2LL * n;             // [n,128] scaled features
  float* bufB = bufA + 128LL * n;         // [n,128] aggregation target
  float* bufC = bufB + 128LL * n;         // [n,128] conv1 hidden
  float* bufD = bufC + 128LL * n;         // [n,64]  conv2 pre-agg
  float* cat  = bufD + 64LL  * n;         // [n,128] concat(h_ori, h_struc)
  float* mlph = bufA;                     // [n,256] overlays bufA+bufB (free by then)

  _Float16* wt1o = (_Float16*)(cat + 128LL * n);  // [128][128]
  _Float16* wt2o = wt1o + 128 * 128;              // [64][128]
  _Float16* wt1s = wt2o + 64 * 128;               // [128][64]
  _Float16* wt2s = wt1s + 128 * 64;               // [64][128]
  _Float16* wtm1 = wt2s + 64 * 128;               // [256][128]
  _Float16* wtm2 = wtm1 + 256 * 128;              // [64][256]

  const int EB = 2048, TB = 256;          // elementwise grid-stride config
  const int AB = 8192;                    // aggregation blocks
  const int WB = 64;                      // weight-transpose blocks

  auto gemm_blocks = [n](int N) { return ((n / 16) * (N / 64) + 3) / 4; };

  // ---- degrees: deg -> rsqrt(clamp(deg,1))
  k_fill<<<EB, TB, 0, stream>>>(dois, 0.0f, 2LL * n);          // dois+diis contiguous
  k_degrees<<<EB, TB, 0, stream>>>(src, dst, dois, diis, e);
  k_rsqrt_clamp<<<(2 * n + TB - 1) / TB, TB, 0, stream>>>(dois, 2 * n);

  // ---- weight pre-transpose (f32 row-major -> f16 [N][K])
  k_w_to_f16t<<<WB, TB, 0, stream>>>(W1o, wt1o, 128, 128);
  k_w_to_f16t<<<WB, TB, 0, stream>>>(W2o, wt2o, 128, 64);
  k_w_to_f16t<<<WB, TB, 0, stream>>>(W1s, wt1s, 64, 128);
  k_w_to_f16t<<<WB, TB, 0, stream>>>(W2s, wt2s, 128, 64);
  k_w_to_f16t<<<WB, TB, 0, stream>>>(Wm1, wtm1, 128, 256);
  k_w_to_f16t<<<WB, TB, 0, stream>>>(Wm2, wtm2, 256, 64);

  // ---- ori conv1 (aggregate-first, K=N=128): h = relu(diis*(Agg(ori*dois))@W1o + b1o)
  k_rowscale<<<EB, TB, 0, stream>>>(ori, dois, bufA, n, 128);
  k_fill<<<EB, TB, 0, stream>>>(bufB, 0.0f, 128LL * n);
  k_aggregate<<<AB, TB, 0, stream>>>(src, dst, bufA, bufB, e, 128);
  k_gemm_wmma<128, false, true, true, true>
      <<<gemm_blocks(128), 128, 0, stream>>>(bufB, wt1o, b1o, nullptr, diis, bufC, n, 128);

  // ---- ori conv2 (W-first, 128->64): cat[:,0:64] = diis*Agg((h*dois)@W2o) + b2o
  k_gemm_wmma<128, true, false, false, false>
      <<<gemm_blocks(64), 128, 0, stream>>>(bufC, wt2o, nullptr, dois, nullptr, bufD, n, 64);
  k_fill<<<EB, TB, 0, stream>>>(bufB, 0.0f, 64LL * n);
  k_aggregate<<<AB, TB, 0, stream>>>(src, dst, bufD, bufB, e, 64);
  k_scale_bias<<<EB, TB, 0, stream>>>(bufB, diis, b2o, cat, n, 64, 128, 0);

  // ---- struc conv1 (aggregate-first, K=64, N=128)
  k_rowscale<<<EB, TB, 0, stream>>>(strc, dois, bufA, n, 64);
  k_fill<<<EB, TB, 0, stream>>>(bufB, 0.0f, 64LL * n);
  k_aggregate<<<AB, TB, 0, stream>>>(src, dst, bufA, bufB, e, 64);
  k_gemm_wmma<64, false, true, true, true>
      <<<gemm_blocks(128), 128, 0, stream>>>(bufB, wt1s, b1s, nullptr, diis, bufC, n, 128);

  // ---- struc conv2 (W-first, 128->64): cat[:,64:128]
  k_gemm_wmma<128, true, false, false, false>
      <<<gemm_blocks(64), 128, 0, stream>>>(bufC, wt2s, nullptr, dois, nullptr, bufD, n, 64);
  k_fill<<<EB, TB, 0, stream>>>(bufB, 0.0f, 64LL * n);
  k_aggregate<<<AB, TB, 0, stream>>>(src, dst, bufD, bufB, e, 64);
  k_scale_bias<<<EB, TB, 0, stream>>>(bufB, diis, b2s, cat, n, 64, 128, 64);

  // ---- MLP: relu(cat@Wm1 + bm1) @ Wm2 + bm2
  k_gemm_wmma<128, false, false, true, true>
      <<<gemm_blocks(256), 128, 0, stream>>>(cat, wtm1, bm1, nullptr, nullptr, mlph, n, 256);
  k_gemm_wmma<256, false, false, true, false>
      <<<gemm_blocks(64), 128, 0, stream>>>(mlph, wtm2, bm2, nullptr, nullptr, out, n, 64);
}
